// Attention_26336739459136
// MI455X (gfx1250) — matrix operand order repaired and checked by reading, not compiled, not run on hardware
//
#include <hip/hip_runtime.h>
#include <hip/hip_bf16.h>

// MI455X / gfx1250 bf16-WMMA attention pipeline (wave32), v4:
// ping-pong double buffering (no register-copy rotation) in mma_2x4 and the S-phase loop.

typedef __bf16 bf16_t;
typedef __attribute__((ext_vector_type(16))) __bf16 v16bf;
typedef __attribute__((ext_vector_type(8)))  float  v8f;

#define HEAD_DIM 1024
#define CDIM     1024
#define NC       8192
#define BS       4096
#define QK_LD    2048

__device__ __forceinline__ bf16_t f2bf(float f) {
  unsigned u = __builtin_bit_cast(unsigned, f);
  u += 0x7fffu + ((u >> 16) & 1u);              // round-to-nearest-even
  unsigned short h = (unsigned short)(u >> 16);
  return __builtin_bit_cast(bf16_t, h);
}

union FragU { v16bf v; uint4 q[2]; };

// A fragment: 16x32 bf16 from row-major M x K (lda elements).
__device__ __forceinline__ v16bf load_a(const bf16_t* p, int lda) {
  int lane = threadIdx.x & 31;
  int m = lane & 15, half = lane >> 4;
  const bf16_t* r = p + m * lda + half * 8;
  FragU f;
  f.q[0] = *(const uint4*)(r);        // K = half*8 + 0..7
  f.q[1] = *(const uint4*)(r + 16);   // K = 16 + half*8 + 0..7
  return f.v;
}

// B fragment: 32x16 bf16 from B^T stored row-major N x K (ldb elements).
__device__ __forceinline__ v16bf load_b(const bf16_t* p, int ldb) {
  int lane = threadIdx.x & 31;
  int n = lane & 15, half = lane >> 4;
  const bf16_t* r = p + n * ldb + half * 8;
  FragU f;
  f.q[0] = *(const uint4*)(r);
  f.q[1] = *(const uint4*)(r + 16);
  return f.v;
}

__device__ __forceinline__ v8f wmma_bf16(v16bf a, v16bf b, v8f c) {
  return __builtin_amdgcn_wmma_f32_16x16x32_bf16(false, a, false, b, (short)0, c,
                                                 false, false);
}

// Pipelined 32x64 tile block over K (K/32 even, >= 4): ping-pong buffers,
// all fragment registers written only by loads (no rotation copies).
__device__ __forceinline__ void mma_2x4(const bf16_t* a0, const bf16_t* a1, int lda,
                                        const bf16_t* b, int ldb, int K,
                                        v8f acc[2][4]) {
  v16bf A0a = load_a(a0, lda), A1a = load_a(a1, lda);
  v16bf Ba[4];
  #pragma unroll
  for (int t = 0; t < 4; ++t) Ba[t] = load_b(b + t * 16 * ldb, ldb);

  for (int d = 0; d < K - 64; d += 64) {
    v16bf A0b = load_a(a0 + d + 32, lda);
    v16bf A1b = load_a(a1 + d + 32, lda);
    v16bf Bb[4];
    #pragma unroll
    for (int t = 0; t < 4; ++t) Bb[t] = load_b(b + t * 16 * ldb + d + 32, ldb);
    #pragma unroll
    for (int t = 0; t < 4; ++t) {              // step d
      acc[0][t] = wmma_bf16(A0a, Ba[t], acc[0][t]);
      acc[1][t] = wmma_bf16(A1a, Ba[t], acc[1][t]);
    }
    A0a = load_a(a0 + d + 64, lda);
    A1a = load_a(a1 + d + 64, lda);
    #pragma unroll
    for (int t = 0; t < 4; ++t) Ba[t] = load_b(b + t * 16 * ldb + d + 64, ldb);
    #pragma unroll
    for (int t = 0; t < 4; ++t) {              // step d+32
      acc[0][t] = wmma_bf16(A0b, Bb[t], acc[0][t]);
      acc[1][t] = wmma_bf16(A1b, Bb[t], acc[1][t]);
    }
  }
  {
    v16bf A0b = load_a(a0 + K - 32, lda);
    v16bf A1b = load_a(a1 + K - 32, lda);
    v16bf Bb[4];
    #pragma unroll
    for (int t = 0; t < 4; ++t) Bb[t] = load_b(b + t * 16 * ldb + K - 32, ldb);
    #pragma unroll
    for (int t = 0; t < 4; ++t) {              // step K-64
      acc[0][t] = wmma_bf16(A0a, Ba[t], acc[0][t]);
      acc[1][t] = wmma_bf16(A1a, Ba[t], acc[1][t]);
    }
    #pragma unroll
    for (int t = 0; t < 4; ++t) {              // step K-32
      acc[0][t] = wmma_bf16(A0b, Bb[t], acc[0][t]);
      acc[1][t] = wmma_bf16(A1b, Bb[t], acc[1][t]);
    }
  }
}

// ---------------- prep kernels ----------------

__global__ __launch_bounds__(256) void k_prep_c(
    const float* __restrict__ s, const float* __restrict__ tg,
    const float* __restrict__ query, bf16_t* __restrict__ c) {
  int i = blockIdx.x * blockDim.x + threadIdx.x;
  c[i] = f2bf(s[i] + tg[i]);
  c[BS * CDIM + i] = f2bf(query[i]);
}

__global__ __launch_bounds__(256) void k_f32_to_bf16(
    const float* __restrict__ in, bf16_t* __restrict__ out) {
  int i = blockIdx.x * blockDim.x + threadIdx.x;
  out[i] = f2bf(in[i]);
}

// out[d][j] = bf16(in[j][d]); in: rows x cols f32
__global__ __launch_bounds__(256) void k_transpose_bf16(
    const float* __restrict__ in, bf16_t* __restrict__ out, int rows, int cols) {
  __shared__ float tile[32][33];
  int bx = blockIdx.x, by = blockIdx.y;
  int tx = threadIdx.x, ty = threadIdx.y;
  int x = bx * 32 + tx;
  #pragma unroll
  for (int i = ty; i < 32; i += 8)
    tile[i][tx] = in[(by * 32 + i) * cols + x];
  __syncthreads();
  int ox = by * 32 + tx;
  #pragma unroll
  for (int i = ty; i < 32; i += 8)
    out[(bx * 32 + i) * rows + ox] = f2bf(tile[tx][i]);
}

// ---------------- QK projection: qk = c @ W[0:2048].T * SCALE ----------------
__global__ __launch_bounds__(256) void k_gemm_qk(
    const bf16_t* __restrict__ c, const bf16_t* __restrict__ w,
    bf16_t* __restrict__ qk, float scale) {
  int wave = threadIdx.x >> 5;
  int gid = blockIdx.x * 8 + wave;
  int row0 = (gid >> 5) * 32;
  int col0 = (gid & 31) * 64;
  v8f acc[2][4] = {};
  mma_2x4(c + row0 * CDIM, c + (row0 + 16) * CDIM, CDIM,
          w + col0 * CDIM, CDIM, CDIM, acc);
  int lane = threadIdx.x & 31;
  int nn = lane & 15, half = lane >> 4;
  #pragma unroll
  for (int rt = 0; rt < 2; ++rt)
    #pragma unroll
    for (int t = 0; t < 4; ++t)
      #pragma unroll
      for (int r = 0; r < 8; ++r)
        qk[(row0 + rt * 16 + r + 8 * half) * QK_LD + col0 + t * 16 + nn] =
            f2bf(acc[rt][t][r] * scale);
}

// ---------------- pass A: partial row stats over one of 8 column slices ------
__global__ __launch_bounds__(256) void k_row_stats_part(
    const bf16_t* __restrict__ qk, float* __restrict__ mpart,
    float* __restrict__ spart) {
  int wave = threadIdx.x >> 5;
  int gid = blockIdx.x * 8 + wave;
  int rowg = gid >> 3;                  // 256 row groups of 32
  int part = gid & 7;                   // 8 column slices of 1024
  int row0 = rowg * 32;
  int jbase = part * 1024;
  int lane = threadIdx.x & 31;
  const bf16_t* qa0 = qk + row0 * QK_LD;
  const bf16_t* qa1 = qk + (row0 + 16) * QK_LD;
  float m[2][8], sum[2][8];
  #pragma unroll
  for (int rt = 0; rt < 2; ++rt)
    #pragma unroll
    for (int r = 0; r < 8; ++r) { m[rt][r] = -3.0e38f; sum[rt][r] = 0.0f; }

  for (int j0 = jbase; j0 < jbase + 1024; j0 += 64) {
    v8f acc[2][4] = {};
    mma_2x4(qa0, qa1, QK_LD, qk + j0 * QK_LD + HEAD_DIM, QK_LD, HEAD_DIM, acc);
    #pragma unroll
    for (int rt = 0; rt < 2; ++rt) {
      #pragma unroll
      for (int r = 0; r < 8; ++r) {
        float v = fmaxf(fmaxf(acc[rt][0][r], acc[rt][1][r]),
                        fmaxf(acc[rt][2][r], acc[rt][3][r]));
        #pragma unroll
        for (int off = 8; off >= 1; off >>= 1)
          v = fmaxf(v, __shfl_xor(v, off, 32));       // 16-lane half-group reduce
        float nm = fmaxf(m[rt][r], v);
        float corr = __expf(m[rt][r] - nm);           // first iter: exp(-huge)=0
        float ss = __expf(acc[rt][0][r] - nm) + __expf(acc[rt][1][r] - nm)
                 + __expf(acc[rt][2][r] - nm) + __expf(acc[rt][3][r] - nm);
        #pragma unroll
        for (int off = 8; off >= 1; off >>= 1)
          ss += __shfl_xor(ss, off, 32);
        sum[rt][r] = sum[rt][r] * corr + ss;
        m[rt][r] = nm;
      }
    }
  }
  if ((lane & 15) == 0) {
    int half = lane >> 4;
    #pragma unroll
    for (int rt = 0; rt < 2; ++rt)
      #pragma unroll
      for (int r = 0; r < 8; ++r) {
        int row = row0 + rt * 16 + r + 8 * half;
        mpart[part * NC + row] = m[rt][r];
        spart[part * NC + row] = sum[rt][r];
      }
  }
}

// merge 8 partial (max,sum) pairs per row -> mrow, rcl
__global__ __launch_bounds__(256) void k_row_stats_merge(
    const float* __restrict__ mpart, const float* __restrict__ spart,
    float* __restrict__ mrow, float* __restrict__ rcl) {
  int i = blockIdx.x * blockDim.x + threadIdx.x;   // 8192 rows
  float m = -3.0e38f;
  #pragma unroll
  for (int p = 0; p < 8; ++p) m = fmaxf(m, mpart[p * NC + i]);
  float sum = 0.0f;
  #pragma unroll
  for (int p = 0; p < 8; ++p)
    sum += spart[p * NC + i] * __expf(mpart[p * NC + i] - m);
  mrow[i] = m;
  rcl[i] = 1.0f / sum;
}

// ---------------- pass B: y = softmax-block @ V ----------------
__global__ __launch_bounds__(256) void k_attn_av(
    const bf16_t* __restrict__ qbase,  // q rows, ld QK_LD
    const bf16_t* __restrict__ kbase,  // k rows (+col offs folded)
    const bf16_t* __restrict__ vt,     // V^T: 1024 x 4096 bf16
    const float* __restrict__ mrow,
    const float* __restrict__ rcl,
    bf16_t* __restrict__ y) {          // 4096 x 1024 bf16
  __shared__ __align__(16) bf16_t Pbuf[32 * 128];
  int wave = threadIdx.x >> 5;
  int lane = threadIdx.x & 31;
  int nn = lane & 15, half = lane >> 4;
  int row0 = blockIdx.x * 32;
  int rt_s = wave >> 2;                    // S-phase row tile
  int jt0  = (wave & 3) * 2;               // S-phase: two of 8 j-tiles
  int d0 = wave * 128;                     // O-phase d slice

  float mv[8], rv[8];
  #pragma unroll
  for (int r = 0; r < 8; ++r) {
    int rr = row0 + rt_s * 16 + r + 8 * half;
    mv[r] = mrow[rr];
    rv[r] = rcl[rr];
  }

  v8f acc[2][8] = {};
  const bf16_t* qa = qbase + (row0 + rt_s * 16) * QK_LD;

  for (int j0 = 0; j0 < 4096; j0 += 128) {
    // --- S phase: 2 tiles per wave, ping-pong pipelined over d ---
    v8f s0 = {}, s1 = {};
    const bf16_t* kb0 = kbase + (j0 + jt0 * 16) * QK_LD;
    const bf16_t* kb1 = kb0 + 16 * QK_LD;
    {
      v16bf Aa  = load_a(qa, QK_LD);
      v16bf B0a = load_b(kb0, QK_LD), B1a = load_b(kb1, QK_LD);
      for (int d = 0; d < HEAD_DIM - 64; d += 64) {
        v16bf Ab  = load_a(qa + d + 32, QK_LD);
        v16bf B0b = load_b(kb0 + d + 32, QK_LD);
        v16bf B1b = load_b(kb1 + d + 32, QK_LD);
        s0 = wmma_bf16(Aa, B0a, s0);
        s1 = wmma_bf16(Aa, B1a, s1);
        Aa  = load_a(qa + d + 64, QK_LD);
        B0a = load_b(kb0 + d + 64, QK_LD);
        B1a = load_b(kb1 + d + 64, QK_LD);
        s0 = wmma_bf16(Ab, B0b, s0);
        s1 = wmma_bf16(Ab, B1b, s1);
      }
      v16bf Ab  = load_a(qa + HEAD_DIM - 32, QK_LD);
      v16bf B0b = load_b(kb0 + HEAD_DIM - 32, QK_LD);
      v16bf B1b = load_b(kb1 + HEAD_DIM - 32, QK_LD);
      s0 = wmma_bf16(Aa, B0a, s0);
      s1 = wmma_bf16(Aa, B1a, s1);
      s0 = wmma_bf16(Ab, B0b, s0);
      s1 = wmma_bf16(Ab, B1b, s1);
    }
    #pragma unroll
    for (int r = 0; r < 8; ++r) {
      int prow = (rt_s * 16 + r + 8 * half) * 128;
      Pbuf[prow + jt0 * 16 + nn]       = f2bf(__expf(s0[r] - mv[r]) * rv[r]);
      Pbuf[prow + (jt0 + 1) * 16 + nn] = f2bf(__expf(s1[r] - mv[r]) * rv[r]);
    }
    __syncthreads();
    // --- O phase: O[32 x 128-slice] += P @ V, V fragments pipelined ---
    #pragma unroll
    for (int ks = 0; ks < 128; ks += 32) {
      #pragma unroll
      for (int rt = 0; rt < 2; ++rt) {
        v16bf Ap = load_a(Pbuf + rt * 16 * 128 + ks, 128);
        v16bf Bv = load_b(vt + d0 * 4096 + j0 + ks, 4096);
        #pragma unroll
        for (int dt = 0; dt < 7; ++dt) {
          v16bf Bvn = load_b(vt + (d0 + (dt + 1) * 16) * 4096 + j0 + ks, 4096);
          acc[rt][dt] = wmma_bf16(Ap, Bv, acc[rt][dt]);
          Bv = Bvn;
        }
        acc[rt][7] = wmma_bf16(Ap, Bv, acc[rt][7]);
      }
    }
    __syncthreads();
  }
  #pragma unroll
  for (int rt = 0; rt < 2; ++rt)
    #pragma unroll
    for (int dt = 0; dt < 8; ++dt)
      #pragma unroll
      for (int r = 0; r < 8; ++r)
        y[(row0 + rt * 16 + r + 8 * half) * HEAD_DIM + d0 + dt * 16 + nn] =
            f2bf(acc[rt][dt][r]);
}

// ---------------- final projection: out = y @ Wp.T (f32 out) ----------------
__global__ __launch_bounds__(256) void k_gemm_proj(
    const bf16_t* __restrict__ ybf, const bf16_t* __restrict__ w,
    float* __restrict__ out) {
  int wave = threadIdx.x >> 5;
  int gid = blockIdx.x * 8 + wave;
  int row0 = (gid >> 4) * 32;
  int col0 = (gid & 15) * 64;
  v8f acc[2][4] = {};
  mma_2x4(ybf + row0 * HEAD_DIM, ybf + (row0 + 16) * HEAD_DIM, HEAD_DIM,
          w + col0 * HEAD_DIM, HEAD_DIM, HEAD_DIM, acc);
  int lane = threadIdx.x & 31;
  int nn = lane & 15, half = lane >> 4;
  #pragma unroll
  for (int rt = 0; rt < 2; ++rt)
    #pragma unroll
    for (int t = 0; t < 4; ++t)
      #pragma unroll
      for (int r = 0; r < 8; ++r)
        out[(row0 + rt * 16 + r + 8 * half) * HEAD_DIM + col0 + t * 16 + nn] =
            acc[rt][t][r];
}

extern "C" void kernel_launch(void* const* d_in, const int* in_sizes, int n_in,
                              void* d_out, int out_size, void* d_ws, size_t ws_size,
                              hipStream_t stream) {
  (void)in_sizes; (void)n_in; (void)out_size; (void)ws_size;
  const float* query = (const float*)d_in[0];
  const float* s     = (const float*)d_in[1];
  const float* tg    = (const float*)d_in[2];
  const float* wqkv  = (const float*)d_in[3];
  const float* wps   = (const float*)d_in[4];
  const float* wpq   = (const float*)d_in[5];
  float* out = (float*)d_out;

  char* ws = (char*)d_ws;
  bf16_t* c_bf   = (bf16_t*)(ws);                 // 8192x1024 bf16
  bf16_t* wqk_bf = (bf16_t*)(ws + 16777216);      // 2048x1024 bf16
  bf16_t* wps_bf = (bf16_t*)(ws + 20971520);      // 1024x1024 bf16
  bf16_t* wpq_bf = (bf16_t*)(ws + 23068672);      // 1024x1024 bf16
  bf16_t* qk     = (bf16_t*)(ws + 25165824);      // 8192x2048 bf16
  bf16_t* qt     = (bf16_t*)(ws + 58720256);      // 1024x4096 bf16 (query^T)
  bf16_t* st     = (bf16_t*)(ws + 67108864);      // 1024x4096 bf16 (s^T)
  float*  mrow   = (float*) (ws + 75497472);      // 8192 f32
  float*  rcl    = (float*) (ws + 75530240);      // 8192 f32
  bf16_t* ys     = (bf16_t*)(ws + 75563008);      // 4096x1024 bf16
  bf16_t* yq     = (bf16_t*)(ws + 83951616);      // 4096x1024 bf16
  float*  mpart  = (float*) (ws + 92340224);      // 8x8192 f32
  float*  spart  = (float*) (ws + 92602368);      // 8x8192 f32  (end ~92.9 MB)

  const float scale = 0.17677669529663687f;       // 1024^-0.25 = 2^-2.5

  k_prep_c<<<(BS * CDIM) / 256, 256, 0, stream>>>(s, tg, query, c_bf);
  k_f32_to_bf16<<<(2048 * 1024) / 256, 256, 0, stream>>>(wqkv, wqk_bf);
  k_f32_to_bf16<<<(1024 * 1024) / 256, 256, 0, stream>>>(wps, wps_bf);
  k_f32_to_bf16<<<(1024 * 1024) / 256, 256, 0, stream>>>(wpq, wpq_bf);
  k_transpose_bf16<<<dim3(1024 / 32, 4096 / 32), dim3(32, 8), 0, stream>>>(query, qt, 4096, 1024);
  k_transpose_bf16<<<dim3(1024 / 32, 4096 / 32), dim3(32, 8), 0, stream>>>(s, st, 4096, 1024);

  k_gemm_qk<<<1024, 256, 0, stream>>>(c_bf, wqk_bf, qk, scale);

  k_row_stats_part<<<256, 256, 0, stream>>>(qk, mpart, spart);
  k_row_stats_merge<<<NC / 256, 256, 0, stream>>>(mpart, spart, mrow, rcl);

  // x_s: attn rows [0,4096), cols [4096,8192), V = query
  k_attn_av<<<128, 256, 0, stream>>>(qk, qk + 4096 * QK_LD + HEAD_DIM,
                                     qt, mrow, rcl, ys);
  // x_q: attn rows [4096,8192), cols [0,4096), V = s
  k_attn_av<<<128, 256, 0, stream>>>(qk + 4096 * QK_LD, qk + HEAD_DIM,
                                     st, mrow + 4096, rcl + 4096, yq);

  k_gemm_proj<<<256, 256, 0, stream>>>(ys, wps_bf, out);
  k_gemm_proj<<<256, 256, 0, stream>>>(yq, wpq_bf, out + 4096 * HEAD_DIM);
}
